// EnhancedVariationalHierarchicalGAT_2413771620930
// MI455X (gfx1250) — compile-verified, matrix operand order checked by reading, not run on hardware
//
#include <hip/hip_runtime.h>
#include <hip/hip_bf16.h>
#include <stdint.h>

typedef __attribute__((ext_vector_type(16))) __bf16 v16bf;
typedef __attribute__((ext_vector_type(8)))  __bf16 v8bf;
typedef __attribute__((ext_vector_type(8)))  float  v8f;

#define NEG_SLOPE 0.2f
#define BN_EPS 1e-5f

__device__ __forceinline__ unsigned short f2bf(float f) {
  unsigned u = __float_as_uint(f);
  u += 0x7FFFu + ((u >> 16) & 1u);          // round-to-nearest-even
  return (unsigned short)(u >> 16);
}
__device__ __forceinline__ unsigned fkey(float f) {   // order-preserving uint key
  unsigned u = __float_as_uint(f);
  return (u & 0x80000000u) ? ~u : (u | 0x80000000u);
}
__device__ __forceinline__ float funkey(unsigned k) {
  return (k & 0x80000000u) ? __uint_as_float(k & 0x7FFFFFFFu) : __uint_as_float(~k);
}
__device__ __forceinline__ void atomAddF(float* p, float v) {
  __hip_atomic_fetch_add(p, v, __ATOMIC_RELAXED, __HIP_MEMORY_SCOPE_AGENT);
}

// ---- CDNA5 async global->LDS copy (ASYNCcnt path), with compile-safe fallback ----
typedef int v4i_vs __attribute__((vector_size(16)));
typedef __attribute__((address_space(1))) v4i_vs* gv4i_p;   // __device__ int4*
typedef __attribute__((address_space(3))) v4i_vs* lv4i_p;   // __shared__ int4*

#if __has_builtin(__builtin_amdgcn_global_load_async_to_lds_b128)
#define HAVE_ASYNC_LDS 1
__device__ __forceinline__ void async_cp16(const void* g, void* l) {
  __builtin_amdgcn_global_load_async_to_lds_b128(
      (gv4i_p)(unsigned long long)g,
      (lv4i_p)(unsigned)(unsigned long long)l, 0, 0);
}
__device__ __forceinline__ void wait_async0() {
#if __has_builtin(__builtin_amdgcn_s_wait_asynccnt)
  __builtin_amdgcn_s_wait_asynccnt(0);
#else
  asm volatile("s_wait_asynccnt 0" ::: "memory");
#endif
}
#else
#define HAVE_ASYNC_LDS 0
__device__ __forceinline__ void async_cp16(const void* g, void* l) {
  *(uint4*)l = *(const uint4*)g;        // synchronous fallback: global load + ds_store
}
__device__ __forceinline__ void wait_async0() {}
#endif

// ---------------- elementwise / fill kernels ----------------
__global__ void k_f32_to_bf16(const float* __restrict__ s, unsigned short* __restrict__ d, long n) {
  long g = (long)blockIdx.x * blockDim.x + threadIdx.x;
  if (g < n) d[g] = f2bf(s[g]);
}
__global__ void k_fill_f32(float* __restrict__ p, float v, long n) {
  long g = (long)blockIdx.x * blockDim.x + threadIdx.x;
  if (g < n) p[g] = v;
}
__global__ void k_fill_u32(unsigned* __restrict__ p, unsigned v, long n) {
  long g = (long)blockIdx.x * blockDim.x + threadIdx.x;
  if (g < n) p[g] = v;
}
__global__ void k_bias_rows256(float* __restrict__ o, const float* __restrict__ bias, long n) {
  long g = (long)blockIdx.x * blockDim.x + threadIdx.x;
  if (g < n) o[g] = bias[(int)(g & 255)];
}
// W[K,Nsrc] (row-major) -> WT[Npad,K] bf16, zero-padded rows
__global__ void k_wtrans(const float* __restrict__ W, unsigned short* __restrict__ WT,
                         int K, int Nsrc, int Npad) {
  long g = (long)blockIdx.x * blockDim.x + threadIdx.x;
  long tot = (long)Npad * K;
  if (g >= tot) return;
  int n = (int)(g / K), k = (int)(g - (long)n * K);
  float v = (n < Nsrc) ? W[(size_t)k * Nsrc + n] : 0.0f;
  WT[(size_t)n * K + k] = f2bf(v);
}

// ---------------- WMMA bf16 GEMM with async-LDS double-buffered staging ----------------
// C[M,Ncols] = A[M,K] * BT[Nld,K]^T + bias
// block = 128 threads (4 waves); block tile = 32 rows x 64 cols
// wave w: two 16x16 accumulators (row halves 0/1) x cols [16w,16w+16), shared B fragment
__global__ __launch_bounds__(128)
void k_gemm_bf16(const unsigned short* __restrict__ A, const unsigned short* __restrict__ BT,
                 const float* __restrict__ bias, float* __restrict__ C,
                 int M, int Ncols, int Nld, int K) {
  __shared__ __align__(16) unsigned short smA[2][32 * 32];  // 32 rows x 32 k (64B/row)
  __shared__ __align__(16) unsigned short smB[2][64 * 32];  // 64 cols x 32 k

  const int t = threadIdx.x;
  const int wave = t >> 5;
  const int lane = t & 31;
  const int r = lane & 15;
  const int khalf = lane >> 4;
  const int rowbase = blockIdx.x * 32;
  const int colbase0 = blockIdx.y * 64;

  auto stage = [&](int buf, int kk) {
    {  // A tile: 32 rows x 64B = 128 x 16B chunks, one per thread
      int row = t >> 2, c = t & 3;
      int gr = rowbase + row; if (gr >= M) gr = M - 1;
      const char* g = (const char*)A + (size_t)gr * ((size_t)K * 2) + (size_t)kk * 2 + c * 16;
      char* l = (char*)&smA[buf][0] + row * 64 + c * 16;
      async_cp16(g, l);
    }
#pragma unroll
    for (int q = 0; q < 2; ++q) {  // B tile: 64 cols x 64B = 256 chunks, two per thread
      int chunk = t + q * 128;
      int col = chunk >> 2, c = chunk & 3;
      int gc = colbase0 + col; if (gc >= Nld) gc = Nld - 1;
      const char* g = (const char*)BT + (size_t)gc * ((size_t)K * 2) + (size_t)kk * 2 + c * 16;
      char* l = (char*)&smB[buf][0] + col * 64 + c * 16;
      async_cp16(g, l);
    }
  };

  v8f acc0 = {}, acc1 = {};
  int buf = 0;
  stage(0, 0);
  for (int kk = 0; kk < K; kk += 32) {
    wait_async0();
    __syncthreads();                       // staged tiles visible to all waves
    if (kk + 32 < K) stage(buf ^ 1, kk + 32);   // overlap next DMA with compute

    // fragments from LDS (16-bit A-matrix 16x32 lane layout):
    // lane needs bytes [khalf*16 .. +15] and [32+khalf*16 .. +15] of its 64B row
    const unsigned short* bp = &smB[buf][(wave * 16 + r) * 32] + khalf * 8;
    v8bf blo = *(const v8bf*)bp;
    v8bf bhi = *(const v8bf*)(bp + 16);
    v16bf bfrag = __builtin_shufflevector(blo, bhi, 0, 1, 2, 3, 4, 5, 6, 7,
                                          8, 9, 10, 11, 12, 13, 14, 15);
    const unsigned short* ap0 = &smA[buf][r * 32] + khalf * 8;
    v8bf a0lo = *(const v8bf*)ap0;
    v8bf a0hi = *(const v8bf*)(ap0 + 16);
    v16bf afrag0 = __builtin_shufflevector(a0lo, a0hi, 0, 1, 2, 3, 4, 5, 6, 7,
                                           8, 9, 10, 11, 12, 13, 14, 15);
    const unsigned short* ap1 = &smA[buf][(16 + r) * 32] + khalf * 8;
    v8bf a1lo = *(const v8bf*)ap1;
    v8bf a1hi = *(const v8bf*)(ap1 + 16);
    v16bf afrag1 = __builtin_shufflevector(a1lo, a1hi, 0, 1, 2, 3, 4, 5, 6, 7,
                                           8, 9, 10, 11, 12, 13, 14, 15);

    acc0 = __builtin_amdgcn_wmma_f32_16x16x32_bf16(false, afrag0, false, bfrag, (short)0, acc0, false, false);
    acc1 = __builtin_amdgcn_wmma_f32_16x16x32_bf16(false, afrag1, false, bfrag, (short)0, acc1, false, false);
    __syncthreads();                       // all waves done reading buf before restage
    buf ^= 1;
  }

  int col = colbase0 + wave * 16 + r;
  if (col < Ncols) {
    float bi = bias ? bias[col] : 0.0f;
#pragma unroll
    for (int v = 0; v < 8; ++v) {          // C/D layout: vgpr v -> M = 8*khalf + v
      int orow = rowbase + khalf * 8 + v;
      if (orow < M) C[(size_t)orow * Ncols + col] = acc0[v] + bi;
      int orow1 = rowbase + 16 + khalf * 8 + v;
      if (orow1 < M) C[(size_t)orow1 * Ncols + col] = acc1[v] + bi;
    }
  }
}

// ---------------- GAT edge phase ----------------
__global__ void k_scores(const float* __restrict__ hpre, const float* __restrict__ asrc,
                         const float* __restrict__ adst, float* __restrict__ als,
                         float* __restrict__ ald, int N) {
  int g = blockIdx.x * blockDim.x + threadIdx.x;
  if (g >= N * 4) return;
  int n = g >> 2, h = g & 3;
  const float4* hp = (const float4*)(hpre + (size_t)n * 256 + h * 64);
  const float4* as = (const float4*)(asrc + h * 64);
  const float4* ad = (const float4*)(adst + h * 64);
  float s0 = 0.f, s1 = 0.f;
#pragma unroll
  for (int i = 0; i < 16; ++i) {
    float4 v = hp[i], a = as[i], b = ad[i];
    s0 += v.x * a.x + v.y * a.y + v.z * a.z + v.w * a.w;
    s1 += v.x * b.x + v.y * b.y + v.z * b.z + v.w * b.w;
  }
  als[g] = s0; ald[g] = s1;
}

__device__ __forceinline__ void edge_sd(const int* ei, int E, int i, int& s, int& d) {
  if (i < E) { s = ei[i]; d = ei[E + i]; } else { s = d = i - E; }  // self-loops appended
}

__global__ void k_edge_max(const int* __restrict__ ei, int E, int N,
                           const float* __restrict__ als, const float* __restrict__ ald,
                           unsigned* __restrict__ mmax) {
  int i = blockIdx.x * blockDim.x + threadIdx.x;
  if (i >= E + N) return;
  int s, d; edge_sd(ei, E, i, s, d);
#pragma unroll
  for (int h = 0; h < 4; ++h) {
    float e = als[s * 4 + h] + ald[d * 4 + h];
    e = e > 0.f ? e : NEG_SLOPE * e;
    atomicMax(&mmax[d * 4 + h], fkey(e));
  }
}

__global__ void k_edge_exp(const int* __restrict__ ei, int E, int N,
                           const float* __restrict__ als, const float* __restrict__ ald,
                           const unsigned* __restrict__ mmax, float* __restrict__ pbuf,
                           float* __restrict__ ssum) {
  int i = blockIdx.x * blockDim.x + threadIdx.x;
  if (i >= E + N) return;
  int s, d; edge_sd(ei, E, i, s, d);
#pragma unroll
  for (int h = 0; h < 4; ++h) {
    float e = als[s * 4 + h] + ald[d * 4 + h];
    e = e > 0.f ? e : NEG_SLOPE * e;
    float pe = expf(e - funkey(mmax[d * 4 + h]));
    pbuf[(size_t)i * 4 + h] = pe;
    atomAddF(&ssum[d * 4 + h], pe);
  }
}

// one thread per (edge, 4 channels): float4 gather of h[src], 4 native f32 atomics to out[dst]
__global__ void k_edge_agg(const int* __restrict__ ei, int E, int N,
                           const float* __restrict__ pbuf, const float* __restrict__ ssum,
                           const float* __restrict__ hpre, float* __restrict__ hout) {
  long g = (long)blockIdx.x * blockDim.x + threadIdx.x;
  long tot = (long)(E + N) * 64;
  if (g >= tot) return;
  int i = (int)(g >> 6);
  int q = (int)(g & 63);
  int j = q * 4;
  int h = q >> 4;
  int s, d; edge_sd(ei, E, i, s, d);
  float alpha = pbuf[(size_t)i * 4 + h] / (ssum[d * 4 + h] + 1e-16f);
  float4 hv = *(const float4*)(hpre + (size_t)s * 256 + j);
  float* o = hout + (size_t)d * 256 + j;
  atomAddF(o + 0, alpha * hv.x);
  atomAddF(o + 1, alpha * hv.y);
  atomAddF(o + 2, alpha * hv.z);
  atomAddF(o + 3, alpha * hv.w);
}

// ---------------- BatchNorm ----------------
__global__ __launch_bounds__(256)
void k_bn_stats(const float* __restrict__ hin, float* __restrict__ sum,
                float* __restrict__ sumsq, int N) {
  int ch = threadIdx.x;
  int base = blockIdx.x * 64;
  int rows = N - base; if (rows > 64) rows = 64;
  float s = 0.f, s2 = 0.f;
  for (int r = 0; r < rows; ++r) {
    float v = hin[(size_t)(base + r) * 256 + ch];
    s += v; s2 += v * v;
  }
  atomAddF(&sum[ch], s);
  atomAddF(&sumsq[ch], s2);
}
__global__ void k_bn_final(const float* __restrict__ sum, const float* __restrict__ sumsq,
                           const float* __restrict__ g, const float* __restrict__ be,
                           float* __restrict__ scale, float* __restrict__ shift, int N) {
  int ch = threadIdx.x;
  float inv = 1.0f / (float)N;
  float mu = sum[ch] * inv;
  float var = sumsq[ch] * inv - mu * mu;
  float sc = g[ch] * rsqrtf(var + BN_EPS);
  scale[ch] = sc;
  shift[ch] = be[ch] - mu * sc;
}
__global__ void k_bn_apply(const float* __restrict__ hin, const float* __restrict__ scale,
                           const float* __restrict__ shift, const float* __restrict__ resid,
                           float* __restrict__ encout, unsigned short* __restrict__ actb, long n) {
  long g = (long)blockIdx.x * blockDim.x + threadIdx.x;
  if (g >= n) return;
  int ch = (int)(g & 255);
  float y = fmaxf(fmaf(hin[g], scale[ch], shift[ch]), 0.0f);
  if (resid) { y += resid[g]; encout[g] = y; }
  actb[g] = f2bf(y);
}

// ---------------- reparameterize ----------------
__global__ void k_z(const float* __restrict__ mu, const float* __restrict__ lv,
                    const float* __restrict__ eps, unsigned short* __restrict__ zb, long n) {
  long g = (long)blockIdx.x * blockDim.x + threadIdx.x;
  if (g >= n) return;
  float z = mu[g] + eps[g] * expf(0.5f * lv[g]);
  zb[g] = f2bf(z);
}

// ---------------- host orchestration ----------------
extern "C" void kernel_launch(void* const* d_in, const int* in_sizes, int n_in,
                              void* d_out, int out_size, void* d_ws, size_t ws_size,
                              hipStream_t stream) {
  (void)n_in; (void)out_size; (void)ws_size;
  const int IN = 256, HC = 256;
  const int N = in_sizes[0] / IN;
  const int E = in_sizes[1] / 2;
  const int Ep = E + N;

  const float* x   = (const float*)d_in[0];
  const int*   ei  = (const int*)d_in[1];
  const float* eps = (const float*)d_in[2];
  const float *Wl[3], *asrc[3], *adst[3], *bl[3], *gl[3], *bel[3];
  for (int l = 0; l < 3; ++l) {
    Wl[l]   = (const float*)d_in[3 + 6 * l];
    asrc[l] = (const float*)d_in[4 + 6 * l];
    adst[l] = (const float*)d_in[5 + 6 * l];
    bl[l]   = (const float*)d_in[6 + 6 * l];
    gl[l]   = (const float*)d_in[7 + 6 * l];
    bel[l]  = (const float*)d_in[8 + 6 * l];
  }
  const float* Wr  = (const float*)d_in[21]; const float* br  = (const float*)d_in[22];
  const float* Wmu = (const float*)d_in[23]; const float* bmu = (const float*)d_in[24];
  const float* Wlv = (const float*)d_in[25]; const float* blv = (const float*)d_in[26];
  const float* Wd  = (const float*)d_in[27]; const float* bd  = (const float*)d_in[28];
  const float* Wa  = (const float*)d_in[29]; const float* ba  = (const float*)d_in[30];

  float* out     = (float*)d_out;                  // [N,32] decoder out
  float* out_mu  = out + (size_t)N * 32;           // [N,32]
  float* out_lv  = out + (size_t)N * 64;           // [N,32]
  float* out_aux = out + (size_t)N * 96;           // [N,30]
  float* out_enc = out + (size_t)N * 96 + (size_t)N * 30;  // [N,256]

  char* wsb = (char*)d_ws;
  size_t off = 0;
  auto carve = [&](size_t bytes) -> void* {
    void* p = wsb + off;
    off += (bytes + 255) & ~(size_t)255;
    return p;
  };
  unsigned short* xb    = (unsigned short*)carve((size_t)N * IN * 2);
  unsigned short* actb  = (unsigned short*)carve((size_t)N * HC * 2);
  float* hpre  = (float*)carve((size_t)N * HC * 4);
  float* hout  = (float*)carve((size_t)N * HC * 4);
  float* resid = (float*)carve((size_t)N * HC * 4);
  float* pbuf  = (float*)carve((size_t)Ep * 4 * 4);
  float* als   = (float*)carve((size_t)N * 4 * 4);
  float* ald   = (float*)carve((size_t)N * 4 * 4);
  unsigned* mmax = (unsigned*)carve((size_t)N * 4 * 4);
  float* ssum  = (float*)carve((size_t)N * 4 * 4);
  float* stats = (float*)carve(4 * HC * 4);        // sum | sumsq | scale | shift
  unsigned short* zb   = (unsigned short*)carve((size_t)N * 32 * 2);
  unsigned short* WT[3];
  for (int l = 0; l < 3; ++l) WT[l] = (unsigned short*)carve((size_t)HC * HC * 2);
  unsigned short* WrT  = (unsigned short*)carve((size_t)HC * HC * 2);
  unsigned short* WmuT = (unsigned short*)carve((size_t)32 * HC * 2);
  unsigned short* WlvT = (unsigned short*)carve((size_t)32 * HC * 2);
  unsigned short* WdT  = (unsigned short*)carve((size_t)32 * 32 * 2);
  unsigned short* WaT  = (unsigned short*)carve((size_t)32 * HC * 2);

  auto blocks = [](long n) { return dim3((unsigned)((n + 255) / 256)); };
  auto gemm = [&](const unsigned short* A, const unsigned short* BT, const float* bias,
                  float* C, int M, int Ncols, int Nld, int K) {
    dim3 grid((M + 31) / 32, (Ncols + 63) / 64);
    k_gemm_bf16<<<grid, 128, 0, stream>>>(A, BT, bias, C, M, Ncols, Nld, K);
  };

  // --- precompute bf16 operands ---
  k_f32_to_bf16<<<blocks((long)N * IN), 256, 0, stream>>>(x, xb, (long)N * IN);
  for (int l = 0; l < 3; ++l)
    k_wtrans<<<blocks((long)HC * HC), 256, 0, stream>>>(Wl[l], WT[l], HC, HC, HC);
  k_wtrans<<<blocks((long)HC * HC), 256, 0, stream>>>(Wr,  WrT,  HC, HC, HC);
  k_wtrans<<<blocks((long)32 * HC), 256, 0, stream>>>(Wmu, WmuT, HC, 32, 32);
  k_wtrans<<<blocks((long)32 * HC), 256, 0, stream>>>(Wlv, WlvT, HC, 32, 32);
  k_wtrans<<<blocks((long)32 * 32), 256, 0, stream>>>(Wd,  WdT,  32, 32, 32);
  k_wtrans<<<blocks((long)32 * HC), 256, 0, stream>>>(Wa,  WaT,  HC, 30, 32);

  // --- residual = x @ Wr + br ---
  gemm(xb, WrT, br, resid, N, HC, HC, IN);

  // --- 3 GAT layers ---
  const unsigned short* act = xb;
  for (int l = 0; l < 3; ++l) {
    gemm(act, WT[l], nullptr, hpre, N, HC, HC, HC);
    k_scores<<<blocks((long)N * 4), 256, 0, stream>>>(hpre, asrc[l], adst[l], als, ald, N);
    k_fill_u32<<<blocks((long)N * 4), 256, 0, stream>>>(mmax, 0u, (long)N * 4);
    k_fill_f32<<<blocks((long)N * 4), 256, 0, stream>>>(ssum, 0.0f, (long)N * 4);
    k_edge_max<<<blocks(Ep), 256, 0, stream>>>(ei, E, N, als, ald, mmax);
    k_edge_exp<<<blocks(Ep), 256, 0, stream>>>(ei, E, N, als, ald, mmax, pbuf, ssum);
    k_bias_rows256<<<blocks((long)N * HC), 256, 0, stream>>>(hout, bl[l], (long)N * HC);
    k_edge_agg<<<blocks((long)Ep * 64), 256, 0, stream>>>(ei, E, N, pbuf, ssum, hpre, hout);
    k_fill_f32<<<blocks(2 * HC), 256, 0, stream>>>(stats, 0.0f, 2 * HC);
    k_bn_stats<<<dim3((N + 63) / 64), 256, 0, stream>>>(hout, stats, stats + HC, N);
    k_bn_final<<<dim3(1), 256, 0, stream>>>(stats, stats + HC, gl[l], bel[l],
                                            stats + 2 * HC, stats + 3 * HC, N);
    bool last = (l == 2);
    k_bn_apply<<<blocks((long)N * HC), 256, 0, stream>>>(
        hout, stats + 2 * HC, stats + 3 * HC,
        last ? resid : (const float*)nullptr,
        last ? out_enc : (float*)nullptr, actb, (long)N * HC);
    act = actb;                                   // actb now holds bf16(enc) after layer 3
  }

  // --- heads ---
  gemm(actb, WmuT, bmu, out_mu, N, 32, 32, HC);
  gemm(actb, WlvT, blv, out_lv, N, 32, 32, HC);
  k_z<<<blocks((long)N * 32), 256, 0, stream>>>(out_mu, out_lv, eps, zb, (long)N * 32);
  gemm(zb, WdT, bd, out, N, 32, 32, 32);
  gemm(actb, WaT, ba, out_aux, N, 30, 32, HC);
}
